// SoftmaxAttention_3882650436316
// MI455X (gfx1250) — compile-verified, matrix operand order checked by reading, not run on hardware
//
#include <hip/hip_runtime.h>
#include <stdint.h>

// ---------------------------------------------------------------------------
// ESIM co-attention for MI455X (gfx1250): fused flash-style, bf16 WMMA core,
// double-buffered async global->LDS K staging (ASYNCcnt pipelining),
// cooperative f32->bf16 conversion, full probability row resident in LDS.
// ---------------------------------------------------------------------------

typedef float   v8f   __attribute__((ext_vector_type(8)));
typedef float   fv4   __attribute__((ext_vector_type(4)));
typedef __bf16  bfv4  __attribute__((ext_vector_type(4)));
typedef __bf16  bfv8  __attribute__((ext_vector_type(8)));
typedef __bf16  bfv16 __attribute__((ext_vector_type(16)));

#define BDIM 64
#define LDIM 1024
#define DDIM 512
#define ROWS 64          // query rows per workgroup
#define NT   (LDIM/16)   // number of 16-column K tiles

#define QP   520         // Q LDS pitch (bf16 elems)
#define PP   1032        // P LDS pitch (bf16 elems)
#define KP   516         // K f32 staging pitch (f32 elems)
#define KBP  520         // K bf16 tile pitch (bf16 elems)
#define VTP  40          // V^T LDS pitch (bf16 elems)

#define QS_OFF 0
#define PS_OFF (QS_OFF + ROWS*QP*2)          // 66,560
#define KV_OFF (PS_OFF + ROWS*PP*2)          // 198,656
#define KVBUFB (16*KP*4)                     // 33,024 per ping-pong buffer
#define KB_OFF (KV_OFF + 2*KVBUFB)           // 264,704
#define CM_OFF (KB_OFF + 16*KBP*2)           // 281,344
#define SMEM_BYTES (CM_OFF + LDIM*4)         // 285,440  (< 320KB WGP LDS)

union BF16x16 { bfv16 v; bfv8 h[2]; };

__device__ __forceinline__ void async_b128_to_lds(uint32_t ldsOff, uint32_t gOff,
                                                  const void* sbase) {
  // GVS mode: 64-bit SGPR base + 32-bit per-lane voffset; VDST = LDS byte addr
  asm volatile("global_load_async_to_lds_b128 %0, %1, %2"
               :: "v"(ldsOff), "v"(gOff), "s"(sbase) : "memory");
}
__device__ __forceinline__ void wait_asynccnt0() {
  asm volatile("s_wait_asynccnt 0" ::: "memory");
}
__device__ __forceinline__ void wait_asynccnt16() {
  // <=16 outstanding: the group issued for tile j+1 may remain in flight,
  // the group for tile j (issued one iteration earlier) has retired (in-order).
  asm volatile("s_wait_asynccnt 16" ::: "memory");
}

// Fragment for v_wmma_f32_16x16x32_bf16, built so each half is one contiguous
// 16B LDS read.  16-bit A layout (ISA 7.12.2): lane L (M=L%16) holds
// K = [kc+hi*8 .. +7] in slots 0-3 and K = [kc+16+hi*8 .. +7] in slots 4-7.
__device__ __forceinline__ bfv16 frag_from_bf16(const __bf16* base, int row,
                                                int pitch, int kc, int hi) {
  const __bf16* p = base + row*pitch + kc + hi*8;
  BF16x16 u;
  u.h[0] = *(const bfv8*)p;
  u.h[1] = *(const bfv8*)(p + 16);
  return u.v;
}
__device__ __forceinline__ v8f wmma_bf16(bfv16 a, bfv16 b, v8f c) {
  return __builtin_amdgcn_wmma_f32_16x16x32_bf16(false, a, false, b,
                                                 (short)0, c, false, false);
}

__global__ __launch_bounds__(128, 1)
void esim_attn_dir(const float* __restrict__ Q, const float* __restrict__ KV,
                   const int* __restrict__ rowMask, const int* __restrict__ colMask,
                   float* __restrict__ Out)
{
  extern __shared__ char smem[];
  __bf16* qs = (__bf16*)(smem + QS_OFF);   // Q tile, bf16 [64][QP]
  __bf16* ps = (__bf16*)(smem + PS_OFF);   // probabilities, bf16 [64][PP]
  float*  kv = (float*)(smem + KV_OFF);    // K f32 staging, 2 x [16][KP]
  __bf16* kb = (__bf16*)(smem + KB_OFF);   // K bf16 tile, [16][KBP]
  __bf16* vt = (__bf16*)(smem + KV_OFF);   // V^T staging [128][VTP] (aliases kv)
  float*  cm = (float*)(smem + CM_OFF);    // column mask, f32 [1024]

  const int tid  = threadIdx.x;
  const int wave = tid >> 5;
  const int lane = tid & 31;
  const int n    = lane & 15;
  const int hi   = lane >> 4;

  const int b  = blockIdx.y;
  const int m0 = blockIdx.x * ROWS;

  const uint32_t kvLds  = (uint32_t)(uintptr_t)kv;  // LDS byte offset (addr[31:0])
  const float* kbase0 = KV + (size_t)b * LDIM * DDIM;

  // issue one 16x512 f32 K tile (tile j) into ping-pong buffer buf
  auto issueK = [&](int j, int buf) {
    const float* src = kbase0 + (size_t)j * 16 * DDIM;
    uint32_t dst = kvLds + (uint32_t)(buf * KVBUFB);
#pragma unroll
    for (int it = 0; it < 16; ++it) {
      int c  = it * 128 + tid;          // float4 id over 16*512/4 = 2048
      int r  = c >> 7;
      int c4 = (c & 127) << 2;
      async_b128_to_lds(dst + (uint32_t)((r * KP + c4) * 4),
                        (uint32_t)((r * DDIM + c4) * 4), src);
    }
  };
  // cooperative f32 -> bf16 conversion of the staged tile into kb
  auto convertK = [&](int buf) {
    const float* src = kv + buf * (KVBUFB / 4);
#pragma unroll
    for (int it = 0; it < 16; ++it) {
      int c  = it * 128 + tid;
      int r  = c >> 7;
      int c4 = (c & 127) << 2;
      fv4 v = *(const fv4*)(src + r * KP + c4);
      bfv4 o;
      o[0] = (__bf16)v[0]; o[1] = (__bf16)v[1];
      o[2] = (__bf16)v[2]; o[3] = (__bf16)v[3];
      *(bfv4*)(kb + r * KBP + c4) = o;          // single ds_store_b64
    }
  };
  // S tile (16x16 per wave) = Q_wave(16x512) x K_tile^T via 16 WMMAs
  auto computeS = [&]() {
    v8f s = {0.f, 0.f, 0.f, 0.f, 0.f, 0.f, 0.f, 0.f};
#pragma unroll 4
    for (int kc = 0; kc < DDIM; kc += 32) {
      bfv16 a  = frag_from_bf16(qs, wave * 16 + n, QP, kc, hi);
      bfv16 bb = frag_from_bf16(kb, n, KBP, kc, hi);
      s = wmma_bf16(a, bb, s);
    }
    return s;
  };

  for (int i = tid; i < LDIM; i += 128)
    cm[i] = (float)colMask[(size_t)b * LDIM + i];

  { // Q tile -> LDS bf16 (coalesced float4 reads)
    const float* qbase = Q + ((size_t)b * LDIM + m0) * DDIM;
    for (int it = 0; it < 64; ++it) {
      int c  = it * 128 + tid;
      int r  = c >> 7;
      int c4 = (c & 127) << 2;
      fv4 v = *(const fv4*)(qbase + (size_t)r * DDIM + c4);
      bfv4 o;
      o[0] = (__bf16)v[0]; o[1] = (__bf16)v[1];
      o[2] = (__bf16)v[2]; o[3] = (__bf16)v[3];
      *(bfv4*)(qs + r * QP + c4) = o;
    }
  }
  __syncthreads();

  // per-lane stats for rows M = wave*16 + hi*8 + i
  float pmax[8], psum[8], inv[8], rmask[8];
#pragma unroll
  for (int i = 0; i < 8; ++i) { pmax[i] = -3.0e38f; psum[i] = 0.f; }
#pragma unroll
  for (int i = 0; i < 8; ++i)
    rmask[i] = (float)rowMask[(size_t)b * LDIM + m0 + wave * 16 + hi * 8 + i];

  // ---------------- pass 1: masked row max (pipelined) ----------------
  issueK(0, 0);
  for (int j = 0; j < NT; ++j) {
    if (j < NT - 1) { issueK(j + 1, (j + 1) & 1); wait_asynccnt16(); }
    else            { wait_asynccnt0(); }
    __syncthreads();                       // kv[j&1] visible to all waves
    convertK(j & 1);
    __syncthreads();                       // kb ready
    v8f s = computeS();
    float c = cm[j * 16 + n];
#pragma unroll
    for (int i = 0; i < 8; ++i) pmax[i] = fmaxf(pmax[i], s[i] * c);
  }
#pragma unroll
  for (int i = 0; i < 8; ++i)
    for (int off = 1; off < 16; off <<= 1)
      pmax[i] = fmaxf(pmax[i], __shfl_xor(pmax[i], off, 16));

  // ---------------- pass 2: P = exp(S-max)*mask -> LDS (pipelined) ----------
  issueK(0, 0);
  for (int j = 0; j < NT; ++j) {
    if (j < NT - 1) { issueK(j + 1, (j + 1) & 1); wait_asynccnt16(); }
    else            { wait_asynccnt0(); }
    __syncthreads();
    convertK(j & 1);
    __syncthreads();
    v8f s = computeS();
    float c = cm[j * 16 + n];
#pragma unroll
    for (int i = 0; i < 8; ++i) {
      float e = (c != 0.f) ? __expf(s[i] * c - pmax[i]) : 0.f;
      psum[i] += e;
      ps[(wave * 16 + hi * 8 + i) * PP + j * 16 + n] = (__bf16)e;
    }
  }
#pragma unroll
  for (int i = 0; i < 8; ++i) {
    for (int off = 1; off < 16; off <<= 1)
      psum[i] += __shfl_xor(psum[i], off, 16);
    inv[i] = rmask[i] / (psum[i] + 1e-13f);   // fold row mask into scale
  }
  __syncthreads();   // vt aliases kv: all staging reads must be done

  // ---------------- stage 3: O = P @ V, normalized ----------------
  for (int dc = 0; dc < DDIM / 128; ++dc) {
    v8f acc[8];
    v8f zero = {0.f, 0.f, 0.f, 0.f, 0.f, 0.f, 0.f, 0.f};
#pragma unroll
    for (int t = 0; t < 8; ++t) acc[t] = zero;

    for (int kc = 0; kc < LDIM; kc += 32) {
      // stage V^T tile: vt[d][k], 128 x 32 bf16 (transposed on the fly)
#pragma unroll
      for (int it = 0; it < 8; ++it) {
        int cI = it * 128 + tid;               // 1024 float4s = 32 rows x 32
        int r  = cI >> 5;
        int c4 = (cI & 31) << 2;
        const float* gsrc = kbase0 + (size_t)(kc + r) * DDIM + dc * 128 + c4;
        fv4 v = *(const fv4*)gsrc;
        if (kc + 32 < LDIM)                     // global_prefetch_b8 next tile
          __builtin_prefetch(gsrc + 32 * DDIM, 0, 1);
        vt[(c4 + 0) * VTP + r] = (__bf16)v[0];
        vt[(c4 + 1) * VTP + r] = (__bf16)v[1];
        vt[(c4 + 2) * VTP + r] = (__bf16)v[2];
        vt[(c4 + 3) * VTP + r] = (__bf16)v[3];
      }
      __syncthreads();

      bfv16 a = frag_from_bf16(ps, wave * 16 + n, PP, kc, hi);
#pragma unroll
      for (int t = 0; t < 8; ++t) {
        bfv16 bb = frag_from_bf16(vt, t * 16 + n, VTP, 0, hi);
        acc[t] = wmma_bf16(a, bb, acc[t]);
      }
      __syncthreads();
    }

    float* obase = Out + ((size_t)b * LDIM + m0 + wave * 16) * DDIM + dc * 128;
#pragma unroll
    for (int t = 0; t < 8; ++t)
#pragma unroll
      for (int i = 0; i < 8; ++i)
        obase[(size_t)(hi * 8 + i) * DDIM + t * 16 + n] = acc[t][i] * inv[i];
  }
}

extern "C" void kernel_launch(void* const* d_in, const int* in_sizes, int n_in,
                              void* d_out, int out_size, void* d_ws, size_t ws_size,
                              hipStream_t stream) {
  (void)in_sizes; (void)n_in; (void)out_size; (void)d_ws; (void)ws_size;
  const float* P  = (const float*)d_in[0];
  const int*   pm = (const int*)d_in[1];
  const float* H  = (const float*)d_in[2];
  const int*   hm = (const int*)d_in[3];
  float* outP = (float*)d_out;                               // attended_premises
  float* outH = outP + (size_t)BDIM * LDIM * DDIM;           // attended_hypotheses

  static_assert(SMEM_BYTES <= 320 * 1024, "LDS over WGP budget");
  (void)hipFuncSetAttribute((const void*)esim_attn_dir,
                            hipFuncAttributeMaxDynamicSharedMemorySize,
                            SMEM_BYTES);

  dim3 grid(LDIM / ROWS, BDIM);
  dim3 blk(128);
  // direction 1: Q=premise, K/V=hypothesis, softmax over h, row-mask=premise
  esim_attn_dir<<<grid, blk, SMEM_BYTES, stream>>>(P, H, pm, hm, outP);
  // direction 2: Q=hypothesis, K/V=premise, softmax over p, row-mask=hypothesis
  esim_attn_dir<<<grid, blk, SMEM_BYTES, stream>>>(H, P, hm, pm, outH);
}
